// RectifiedLinearAttention_22333829939799
// MI455X (gfx1250) — matrix operand order repaired and checked by reading, not compiled, not run on hardware
//
#include <hip/hip_runtime.h>

// ---------------- types ----------------
typedef unsigned short u16;
typedef __attribute__((ext_vector_type(4)))  u16    u16x4;
typedef __attribute__((ext_vector_type(8)))  u16    u16x8;
typedef __attribute__((ext_vector_type(16))) __bf16 v16bf;
typedef __attribute__((ext_vector_type(8)))  float  v8f;

union BfOp {
    v16bf bf;
    u16x8 h[2];
    u16   u[16];
};

// Block all instruction motion across this point (pins software pipeline).
#define SCHED_FENCE() __builtin_amdgcn_sched_barrier(0)

__device__ __forceinline__ u16 f2bf(float f) {
    unsigned u = __float_as_uint(f);
    u += 0x7FFFu + ((u >> 16) & 1u);   // round-to-nearest-even
    return (u16)(u >> 16);
}

__device__ __forceinline__ v8f wmma_bf16(const BfOp& a, const BfOp& b, v8f c) {
    return __builtin_amdgcn_wmma_f32_16x16x32_bf16(
        /*neg_a=*/false, a.bf, /*neg_b=*/false, b.bf,
        /*c_mod=*/(short)0, c, /*reuse_a=*/false, /*reuse_b=*/false);
}

// ---------------- problem constants ----------------
#define BN      2
#define SEQ     4096
#define DIMC    512
#define HEADS   8
#define DH      64
#define MTOT    (BN * SEQ)           // 8192
#define QKVN    (3 * DIMC)           // 1536
#define SCALE_Q 0.125f               // 64^-0.5

// ---------------- kernel 0a: f32 -> bf16 row-major copy ----------------
__global__ __launch_bounds__(256) void cvt_rowmajor(const float* __restrict__ src,
                                                    u16* __restrict__ dst, int n) {
    int i = blockIdx.x * 256 + threadIdx.x;
    if (i < n) dst[i] = f2bf(src[i]);
}

// ---------------- kernel 0b: f32 [K x N] -> bf16 transposed [N x K] ----------------
__global__ __launch_bounds__(256) void cvt_transpose(const float* __restrict__ src,
                                                     u16* __restrict__ dst,
                                                     int Nout, int Kin) {
    int i = blockIdx.x * 256 + threadIdx.x;
    if (i < Nout * Kin) {
        int n = i / Kin, k = i % Kin;
        dst[i] = f2bf(src[(size_t)k * Nout + n]);
    }
}

// ---------------- shared GEMM tile loaders (A row-major, B = transposed weights) ----------------
__device__ __forceinline__ void gemm_ldA(const u16* __restrict__ arow, int k0, int lh, BfOp& a) {
    a.h[0] = *(const u16x8*)(arow + k0 + lh * 16);
    a.h[1] = *(const u16x8*)(arow + k0 + lh * 16 + 8);
}
__device__ __forceinline__ void gemm_ldB(const u16* __restrict__ bcol, int k0, int lh, BfOp* b) {
#pragma unroll
    for (int t = 0; t < 4; ++t) {
        const u16* p = bcol + (size_t)t * 16 * DIMC + k0 + lh * 16;
        b[t].h[0] = *(const u16x8*)(p);
        b[t].h[1] = *(const u16x8*)(p + 8);
    }
}

// ---------------- kernel 1: qkv = x @ w_qkv, scatter to head layout ----------------
// grid(24, 128), block 128 (4 waves). Wave: 16 rows x 64 cols, K=512,
// register double-buffered with sched fences so prefetch loads stay hoisted.
__global__ __launch_bounds__(128) void qkv_gemm(const u16* __restrict__ xbf,
                                                const u16* __restrict__ wT,
                                                u16* __restrict__ qws,
                                                u16* __restrict__ kws,
                                                u16* __restrict__ vTws) {
    const int lane = threadIdx.x & 31;
    const int wave = threadIdx.x >> 5;
    const int lr = lane & 15, lh = lane >> 4;
    const int r0 = (blockIdx.y * 4 + wave) * 16;
    const int c0 = blockIdx.x * 64;

    const u16* arow = xbf + (size_t)(r0 + lr) * DIMC;
    const u16* bcol = wT + (size_t)(c0 + lr) * DIMC;

    v8f acc[4] = {};
    BfOp aA, bA[4], aB, bB[4];
    gemm_ldA(arow, 0, lh, aA);
    gemm_ldB(bcol, 0, lh, bA);
    for (int k0 = 0; k0 < DIMC; k0 += 64) {
        SCHED_FENCE();
        gemm_ldA(arow, k0 + 32, lh, aB);
        gemm_ldB(bcol, k0 + 32, lh, bB);
        SCHED_FENCE();
#pragma unroll
        for (int t = 0; t < 4; ++t) acc[t] = wmma_bf16(aA, bA[t], acc[t]);
        SCHED_FENCE();
        const int kn = (k0 + 64 < DIMC) ? k0 + 64 : 0;
        gemm_ldA(arow, kn, lh, aA);
        gemm_ldB(bcol, kn, lh, bA);
        SCHED_FENCE();
#pragma unroll
        for (int t = 0; t < 4; ++t) acc[t] = wmma_bf16(aB, bB[t], acc[t]);
        SCHED_FENCE();
    }

    const int which = c0 >> 9;            // 0=q, 1=k, 2=v (uniform per block)
    const int head  = (c0 & 511) >> 6;    // uniform per block
    const int bidx  = r0 >> 12;
    const size_t bh = (size_t)bidx * HEADS + head;
#pragma unroll
    for (int t = 0; t < 4; ++t) {
        const int d = t * 16 + lr;
#pragma unroll
        for (int v = 0; v < 8; ++v) {
            const int n = (r0 + lh * 8 + v) & (SEQ - 1);
            float val = acc[t][v];
            if (which == 0)      qws[(bh * SEQ + n) * DH + d] = f2bf(val * SCALE_Q);
            else if (which == 1) kws[(bh * SEQ + n) * DH + d] = f2bf(val);
            else                 vTws[(bh * DH + d) * SEQ + n] = f2bf(val);  // V transposed
        }
    }
}

// ---------------- kernel 2: fused relu attention, no LDS bounce ----------------
// Computes S^T = K_tile @ Q^T; the C-layout of S^T equals the A-operand layout
// required for O += relu(S) @ V, so relu+cvt happens entirely in registers.
// grid(32, 16), block 256 (8 waves). Wave: 16 q-rows x 64 d-cols over 4096 keys.
__device__ __forceinline__ void attn_ldK(const u16* __restrict__ kbase, int j0,
                                         int lr, int lh, BfOp* ak /*[mj*2+kh]*/) {
#pragma unroll
    for (int mj = 0; mj < 2; ++mj) {
        const u16* kr = kbase + (size_t)(j0 + mj * 16 + lr) * DH;
#pragma unroll
        for (int kh = 0; kh < 2; ++kh) {
            ak[mj * 2 + kh].h[0] = *(const u16x8*)(kr + kh * 32 + lh * 8);
            ak[mj * 2 + kh].h[1] = *(const u16x8*)(kr + kh * 32 + 16 + lh * 8);
        }
    }
}
__device__ __forceinline__ void attn_ldV(const u16* __restrict__ vbase, int j0,
                                         int lr, int lh, BfOp* av) {
#pragma unroll
    for (int t = 0; t < 4; ++t) {
        const u16* vr = vbase + (size_t)(t * 16 + lr) * SEQ + j0 + lh * 8;
        av[t].h[0] = *(const u16x8*)(vr);
        av[t].h[1] = *(const u16x8*)(vr + 16);
    }
}
__device__ __forceinline__ void attn_step(const BfOp* ak, const BfOp* av,
                                          const BfOp* bq, v8f* accO) {
    // S^T (32 keys x 16 queries) as two 16x16 C tiles
    v8f st0 = {}, st1 = {};
    st0 = wmma_bf16(ak[0], bq[0], st0);
    st0 = wmma_bf16(ak[1], bq[1], st0);
    st1 = wmma_bf16(ak[2], bq[0], st1);
    st1 = wmma_bf16(ak[3], bq[1], st1);
    // relu + cvt directly into GEMM2 A-operand layout (register-only, shuffle-free)
    BfOp a2;
#pragma unroll
    for (int e = 0; e < 8; ++e) a2.u[e]     = f2bf(fmaxf(st0[e], 0.0f));
#pragma unroll
    for (int e = 0; e < 8; ++e) a2.u[8 + e] = f2bf(fmaxf(st1[e], 0.0f));
#pragma unroll
    for (int t = 0; t < 4; ++t) accO[t] = wmma_bf16(a2, av[t], accO[t]);
}

__global__ __launch_bounds__(256) void relu_attn(const u16* __restrict__ qws,
                                                 const u16* __restrict__ kws,
                                                 const u16* __restrict__ vTws,
                                                 float* __restrict__ attn_out) {
    const int lane = threadIdx.x & 31;
    const int wave = threadIdx.x >> 5;
    const int lr = lane & 15, lh = lane >> 4;
    const int bh = blockIdx.y;
    const int n0 = blockIdx.x * 128 + wave * 16;

    // Q as B-operand (Q^T): lane = query col, elements = two 8-wide d runs at 8*lh and 16+8*lh
    const u16* qrow = qws + ((size_t)bh * SEQ + n0 + lr) * DH;
    BfOp bq[2];
#pragma unroll
    for (int kh = 0; kh < 2; ++kh) {
        bq[kh].h[0] = *(const u16x8*)(qrow + kh * 32 + lh * 8);
        bq[kh].h[1] = *(const u16x8*)(qrow + kh * 32 + 16 + lh * 8);
    }

    const u16* kbase = kws  + (size_t)bh * SEQ * DH;
    const u16* vbase = vTws + (size_t)bh * DH * SEQ;

    v8f accO[4] = {};
    BfOp akA[4], avA[4], akB[4], avB[4];
    attn_ldK(kbase, 0, lr, lh, akA);
    attn_ldV(vbase, 0, lr, lh, avA);
    for (int j0 = 0; j0 < SEQ; j0 += 64) {
        SCHED_FENCE();
        attn_ldK(kbase, j0 + 32, lr, lh, akB);   // prefetch tile i+1
        attn_ldV(vbase, j0 + 32, lr, lh, avB);
        SCHED_FENCE();
        attn_step(akA, avA, bq, accO);           // compute tile i
        SCHED_FENCE();
        const int jn = (j0 + 64 < SEQ) ? j0 + 64 : 0;   // harmless wrap prefetch
        attn_ldK(kbase, jn, lr, lh, akA);        // prefetch tile i+2
        attn_ldV(vbase, jn, lr, lh, avA);
        SCHED_FENCE();
        attn_step(akB, avB, bq, accO);           // compute tile i+1
        SCHED_FENCE();
    }

    // write [b, n, h*64+d] f32 so layernorm reads contiguous rows
    const int bidx = bh >> 3, h = bh & 7;
#pragma unroll
    for (int t = 0; t < 4; ++t) {
        const int col = h * DH + t * 16 + lr;
#pragma unroll
        for (int v = 0; v < 8; ++v) {
            const int n = n0 + lh * 8 + v;
            attn_out[((size_t)bidx * SEQ + n) * DIMC + col] = accO[t][v];
        }
    }
}

// ---------------- kernel 3: layernorm (one wave32 per row), bf16 output ----------------
__global__ __launch_bounds__(256) void layernorm_k(const float* __restrict__ src,
                                                   const float* __restrict__ gamma,
                                                   const float* __restrict__ beta,
                                                   u16* __restrict__ dst) {
    const int lane = threadIdx.x & 31;
    const int wave = threadIdx.x >> 5;
    const size_t row = (size_t)blockIdx.x * 8 + wave;
    const float* p = src + row * DIMC;

    float4 x[4];
    float sum = 0.f, sq = 0.f;
#pragma unroll
    for (int w = 0; w < 4; ++w) {
        x[w] = *(const float4*)(p + w * 128 + lane * 4);
        sum += x[w].x + x[w].y + x[w].z + x[w].w;
        sq  += x[w].x * x[w].x + x[w].y * x[w].y + x[w].z * x[w].z + x[w].w * x[w].w;
    }
#pragma unroll
    for (int off = 16; off > 0; off >>= 1) {
        sum += __shfl_xor(sum, off, 32);
        sq  += __shfl_xor(sq, off, 32);
    }
    const float mu  = sum * (1.0f / DIMC);
    const float var = sq * (1.0f / DIMC) - mu * mu;
    const float rs  = rsqrtf(var + 1e-5f);

    u16* o = dst + row * DIMC;
#pragma unroll
    for (int w = 0; w < 4; ++w) {
        const int c = w * 128 + lane * 4;
        float t0[4] = { x[w].x, x[w].y, x[w].z, x[w].w };
        u16x4 r;
#pragma unroll
        for (int e = 0; e < 4; ++e)
            r[e] = f2bf((t0[e] - mu) * rs * gamma[c + e] + beta[c + e]);
        *(u16x4*)(o + c) = r;
    }
}

// ---------------- kernel 4: out = ln @ w_out + b_out (f32 result) ----------------
// grid(8, 128), block 128 (4 waves), double-buffered with sched fences.
__global__ __launch_bounds__(128) void out_gemm(const u16* __restrict__ abf,
                                                const u16* __restrict__ wT,
                                                const float* __restrict__ bias,
                                                float* __restrict__ out) {
    const int lane = threadIdx.x & 31;
    const int wave = threadIdx.x >> 5;
    const int lr = lane & 15, lh = lane >> 4;
    const int r0 = (blockIdx.y * 4 + wave) * 16;
    const int c0 = blockIdx.x * 64;

    const u16* arow = abf + (size_t)(r0 + lr) * DIMC;
    const u16* bcol = wT + (size_t)(c0 + lr) * DIMC;

    v8f acc[4] = {};
    BfOp aA, bA[4], aB, bB[4];
    gemm_ldA(arow, 0, lh, aA);
    gemm_ldB(bcol, 0, lh, bA);
    for (int k0 = 0; k0 < DIMC; k0 += 64) {
        SCHED_FENCE();
        gemm_ldA(arow, k0 + 32, lh, aB);
        gemm_ldB(bcol, k0 + 32, lh, bB);
        SCHED_FENCE();
#pragma unroll
        for (int t = 0; t < 4; ++t) acc[t] = wmma_bf16(aA, bA[t], acc[t]);
        SCHED_FENCE();
        const int kn = (k0 + 64 < DIMC) ? k0 + 64 : 0;
        gemm_ldA(arow, kn, lh, aA);
        gemm_ldB(bcol, kn, lh, bA);
        SCHED_FENCE();
#pragma unroll
        for (int t = 0; t < 4; ++t) acc[t] = wmma_bf16(aB, bB[t], acc[t]);
        SCHED_FENCE();
    }
#pragma unroll
    for (int t = 0; t < 4; ++t) {
        const int col = c0 + t * 16 + lr;
        const float bv = bias[col];
#pragma unroll
        for (int v = 0; v < 8; ++v)
            out[(size_t)(r0 + lh * 8 + v) * DIMC + col] = acc[t][v] + bv;
    }
}

// ---------------- launcher ----------------
extern "C" void kernel_launch(void* const* d_in, const int* in_sizes, int n_in,
                              void* d_out, int out_size, void* d_ws, size_t ws_size,
                              hipStream_t stream) {
    const float* x        = (const float*)d_in[0];
    const float* w_qkv    = (const float*)d_in[1];
    const float* ln_gamma = (const float*)d_in[2];
    const float* ln_beta  = (const float*)d_in[3];
    const float* w_out    = (const float*)d_in[4];
    const float* b_out    = (const float*)d_in[5];
    float* out = (float*)d_out;

    // workspace layout (byte offsets, all >=16B aligned)
    char* ws = (char*)d_ws;
    constexpr size_t off_xbf   = 0;                                    // 8192*512 bf16
    constexpr size_t off_wqkvT = off_xbf   + (size_t)MTOT * DIMC * 2;  // 1536*512 bf16
    constexpr size_t off_woutT = off_wqkvT + (size_t)QKVN * DIMC * 2;  // 512*512 bf16
    constexpr size_t off_q     = off_woutT + (size_t)DIMC * DIMC * 2;  // 16*4096*64 bf16
    constexpr size_t off_k     = off_q     + (size_t)BN * HEADS * SEQ * DH * 2;
    constexpr size_t off_vT    = off_k     + (size_t)BN * HEADS * SEQ * DH * 2;
    constexpr size_t off_attn  = off_vT    + (size_t)BN * HEADS * SEQ * DH * 2; // f32
    constexpr size_t off_ln    = off_attn  + (size_t)MTOT * DIMC * 4;  // bf16

    u16*   xbf   = (u16*)(ws + off_xbf);
    u16*   wqkvT = (u16*)(ws + off_wqkvT);
    u16*   woutT = (u16*)(ws + off_woutT);
    u16*   qws   = (u16*)(ws + off_q);
    u16*   kws   = (u16*)(ws + off_k);
    u16*   vTws  = (u16*)(ws + off_vT);
    float* attn  = (float*)(ws + off_attn);
    u16*   lnbf  = (u16*)(ws + off_ln);

    // 0) precision conversions (weights transposed for contiguous B-operand loads)
    cvt_rowmajor<<<(MTOT * DIMC + 255) / 256, 256, 0, stream>>>(x, xbf, MTOT * DIMC);
    cvt_transpose<<<(QKVN * DIMC + 255) / 256, 256, 0, stream>>>(w_qkv, wqkvT, QKVN, DIMC);
    cvt_transpose<<<(DIMC * DIMC + 255) / 256, 256, 0, stream>>>(w_out, woutT, DIMC, DIMC);

    // 1) qkv projection -> per-head q (pre-scaled), k, v^T
    qkv_gemm<<<dim3(QKVN / 64, MTOT / 64), 128, 0, stream>>>(xbf, wqkvT, qws, kws, vTws);

    // 2) fused relu attention (never materializes the 4096x4096 score matrix)
    relu_attn<<<dim3(SEQ / 128, BN * HEADS), 256, 0, stream>>>(qws, kws, vTws, attn);

    // 3) layernorm -> bf16
    layernorm_k<<<MTOT / 8, 256, 0, stream>>>(attn, ln_gamma, ln_beta, lnbf);

    // 4) output projection + bias -> f32 d_out
    out_gemm<<<dim3(DIMC / 64, MTOT / 64), 128, 0, stream>>>(lnbf, woutT, b_out, out);
}
